// XrayNet_14302241095787
// MI455X (gfx1250) — matrix operand order repaired and checked by reading, not compiled, not run on hardware
//
#include <hip/hip_runtime.h>
#include <hip/hip_bf16.h>

// ---------------------------------------------------------------------------
// Problem constants
// ---------------------------------------------------------------------------
static constexpr int kB = 64;      // batch
static constexpr int kL = 256;     // sequence length (tokens used: 0..254)
static constexpr int kT = 255;     // timesteps
static constexpr int kH = 1024;    // hidden
static constexpr int kE = 512;     // embedding dim
static constexpr int kV = 8000;    // vocab

typedef __attribute__((ext_vector_type(16))) __bf16 v16bf;
typedef __attribute__((ext_vector_type(8)))  __bf16 v8bf;
typedef __attribute__((ext_vector_type(8)))  float  v8f;

// f32 -> bf16, round-to-nearest-even
__device__ __forceinline__ unsigned short f2bf(float x) {
  unsigned u = __float_as_uint(x);
  u += 0x7fffu + ((u >> 16) & 1u);
  return (unsigned short)(u >> 16);
}

__device__ __forceinline__ float sigmoidf_(float x) {
  return 1.0f / (1.0f + expf(-x));
}

// Build a 16x32 bf16 A-fragment for this lane from a row-major bf16 row.
// Lane l (l<16 -> K runs [0,8)+[16,24); l>=16 -> [8,16)+[24,32) within kbase).
__device__ __forceinline__ v16bf load_a_frag(const unsigned short* __restrict__ row,
                                             int kbase, int hi8) {
  const v8bf lo = *(const v8bf*)(row + kbase + hi8);
  const v8bf hi = *(const v8bf*)(row + kbase + hi8 + 16);
  return __builtin_shufflevector(lo, hi, 0, 1, 2, 3, 4, 5, 6, 7,
                                 8, 9, 10, 11, 12, 13, 14, 15);
}

#define WMMA_BF16(A, Bm, C) \
  __builtin_amdgcn_wmma_f32_16x16x32_bf16(false, (A), false, (Bm), (short)0, (C), false, false)

// ---------------------------------------------------------------------------
// Software-pipelined 16xN GEMM segment: one shared A stream, four B streams,
// four f32 accumulators. Loads for step kt are issued BEFORE the WMMAs of
// step kt-1 so the scheduler can keep 5 loads in flight per WMMA group and
// issue partial s_wait_loadcnt instead of full drains.
// bpN = B-fragment base for accumulator N (already offset by lane*16);
// consecutive k-steps are 512 bf16 apart.
// ---------------------------------------------------------------------------
__device__ __forceinline__ void gemm4(const unsigned short* __restrict__ arow,
                                      const unsigned short* __restrict__ bp0,
                                      const unsigned short* __restrict__ bp1,
                                      const unsigned short* __restrict__ bp2,
                                      const unsigned short* __restrict__ bp3,
                                      int nKT, int hi8, v8f (&acc)[4]) {
  v16bf a  = load_a_frag(arow, 0, hi8);
  v16bf b0 = *(const v16bf*)(bp0);
  v16bf b1 = *(const v16bf*)(bp1);
  v16bf b2 = *(const v16bf*)(bp2);
  v16bf b3 = *(const v16bf*)(bp3);
  for (int kt = 1; kt < nKT; ++kt) {
    const size_t o = (size_t)kt << 9;           // kt*512 bf16
    const v16bf an = load_a_frag(arow, kt << 5, hi8);
    const v16bf c0 = *(const v16bf*)(bp0 + o);
    const v16bf c1 = *(const v16bf*)(bp1 + o);
    const v16bf c2 = *(const v16bf*)(bp2 + o);
    const v16bf c3 = *(const v16bf*)(bp3 + o);
    acc[0] = WMMA_BF16(a, b0, acc[0]);
    acc[1] = WMMA_BF16(a, b1, acc[1]);
    acc[2] = WMMA_BF16(a, b2, acc[2]);
    acc[3] = WMMA_BF16(a, b3, acc[3]);
    a = an; b0 = c0; b1 = c1; b2 = c2; b3 = c3;
  }
  acc[0] = WMMA_BF16(a, b0, acc[0]);
  acc[1] = WMMA_BF16(a, b1, acc[1]);
  acc[2] = WMMA_BF16(a, b2, acc[2]);
  acc[3] = WMMA_BF16(a, b3, acc[3]);
}

// ---------------------------------------------------------------------------
// Prep kernels
// ---------------------------------------------------------------------------
__global__ void cvt_bf16_kernel(const float* __restrict__ src,
                                unsigned short* __restrict__ dst, int n) {
  int i = blockIdx.x * blockDim.x + threadIdx.x;
  if (i < n) dst[i] = f2bf(src[i]);
}

// Pack a row-major (N,K) f32 weight into bf16 WMMA B-fragments:
// frag = nt*(K/32)+kt ; dst[frag*512 + l*16 + e] = W[(nt*16+(l&15))*K + kt*32 + (l>=16?8:0) + (e&7) + (e>=8?16:0)]
__global__ void pack_w_kernel(const float* __restrict__ W,
                              unsigned short* __restrict__ dst, int N, int K) {
  long long tid = (long long)blockIdx.x * blockDim.x + threadIdx.x;
  if (tid >= (long long)N * K) return;
  int e = (int)(tid & 15);
  int l = (int)((tid >> 4) & 31);
  long long frag = tid >> 9;
  int nKT = K >> 5;
  int kt = (int)(frag % nKT);
  int nt = (int)(frag / nKT);
  int n = nt * 16 + (l & 15);
  int k = (kt << 5) + ((l >> 4) << 3) + (e & 7) + ((e >> 3) << 4);
  dst[tid] = f2bf(W[(size_t)n * K + k]);
}

__global__ void bsum_kernel(const float* __restrict__ a, const float* __restrict__ b,
                            float* __restrict__ dst, int n) {
  int i = blockIdx.x * blockDim.x + threadIdx.x;
  if (i < n) dst[i] = a[i] + b[i];
}

__global__ void init_state_kernel(const float* __restrict__ h0, const float* __restrict__ c0,
                                  unsigned short* __restrict__ h1, unsigned short* __restrict__ h2,
                                  float* __restrict__ c1, float* __restrict__ c2, int n) {
  int i = blockIdx.x * blockDim.x + threadIdx.x;
  if (i < n) {
    h1[i] = f2bf(h0[i]);
    c1[i] = c0[i];
    h2[i] = 0;        // bf16(0)
    c2[i] = 0.0f;
  }
}

// ---------------------------------------------------------------------------
// Fused LSTM cell kernel: gates = X@Wx^T + Hprev@Wh^T + bsum, then cell update.
// Grid: (8, 4), block 256 (8 waves). Wave w handles hidden tile j = bx*8+w
// (16 hidden units) for batch tile mt = by, computing ALL FOUR gate tiles so
// the A fragment is reused 4x and the pointwise LSTM update is in-register.
// ---------------------------------------------------------------------------
template <bool HAS_IDX>
__global__ void lstm_cell_kernel(const unsigned short* __restrict__ X, // bf16 (rows x K1)
                                 const int* __restrict__ idx,          // tokens (stride kL) or null
                                 int K1,
                                 const unsigned short* __restrict__ Wx, // packed frags (4H/16, K1/32)
                                 const unsigned short* __restrict__ Hprev, // bf16 (B, K2)
                                 int K2,
                                 const unsigned short* __restrict__ Wh,
                                 const float* __restrict__ bsum, // 4H
                                 float* __restrict__ c,          // (B,H) in-place
                                 unsigned short* __restrict__ Hout) { // bf16 (B,H)
  const int lane = threadIdx.x & 31;
  const int wave = threadIdx.x >> 5;
  const int j = blockIdx.x * 8 + wave;   // hidden tile 0..63
  const int mt = blockIdx.y;             // batch tile 0..3
  const int col = lane & 15;
  const int hi8 = (lane >> 4) << 3;      // 0 or 8
  const int m = mt * 16 + col;           // A-matrix row for this lane

  const unsigned short* arow1;
  if (HAS_IDX) {
    const int tok = idx[(size_t)m * kL];
    arow1 = X + (size_t)tok * K1;
  } else {
    arow1 = X + (size_t)m * K1;
  }
  const unsigned short* arow2 = Hprev + (size_t)m * K2;

  v8f acc[4] = {};
  const int nKT1 = K1 >> 5;
  const int nKT2 = K2 >> 5;

  // X @ Wx^T
  gemm4(arow1,
        Wx + ((size_t)(0 * 64 + j) * nKT1 << 9) + lane * 16,
        Wx + ((size_t)(1 * 64 + j) * nKT1 << 9) + lane * 16,
        Wx + ((size_t)(2 * 64 + j) * nKT1 << 9) + lane * 16,
        Wx + ((size_t)(3 * 64 + j) * nKT1 << 9) + lane * 16,
        nKT1, hi8, acc);
  // + Hprev @ Wh^T
  gemm4(arow2,
        Wh + ((size_t)(0 * 64 + j) * nKT2 << 9) + lane * 16,
        Wh + ((size_t)(1 * 64 + j) * nKT2 << 9) + lane * 16,
        Wh + ((size_t)(2 * 64 + j) * nKT2 << 9) + lane * 16,
        Wh + ((size_t)(3 * 64 + j) * nKT2 << 9) + lane * 16,
        nKT2, hi8, acc);

  // Pointwise cell update; C/D layout: lane -> column, VGPR r -> row (+8 for lanes>=16)
  const int n = j * 16 + col;             // hidden unit
  const float bi = bsum[0 * kH + n];
  const float bf = bsum[1 * kH + n];
  const float bg = bsum[2 * kH + n];
  const float bo = bsum[3 * kH + n];
#pragma unroll
  for (int r = 0; r < 8; ++r) {
    const int row = mt * 16 + hi8 + r;    // batch index
    const float iv = sigmoidf_(acc[0][r] + bi);
    const float fv = sigmoidf_(acc[1][r] + bf);
    const float gv = tanhf(acc[2][r] + bg);
    const float ov = sigmoidf_(acc[3][r] + bo);
    const size_t off = (size_t)row * kH + n;
    const float cn = fv * c[off] + iv * gv;
    c[off] = cn;
    Hout[off] = f2bf(ov * tanhf(cn));
  }
}

// ---------------------------------------------------------------------------
// Output projection: logits = H2 @ Wout^T + b_out, written raw into d_out.
// Each wave owns 4 consecutive vocab tiles so the A fragment (shared h2
// row-block) is loaded once per k-step and reused by 4 WMMAs.
// Grid: (16, 4), block 256 (8 waves); wave job in [0,125), j0 = job*4.
// ---------------------------------------------------------------------------
__global__ void logits_kernel(const unsigned short* __restrict__ H2, // bf16 (B,H)
                              const unsigned short* __restrict__ Wpk, // packed (V/16, H/32)
                              const float* __restrict__ bout,
                              float* __restrict__ out, int t) {
  const int lane = threadIdx.x & 31;
  const int wave = threadIdx.x >> 5;
  const int job = blockIdx.x * 8 + wave;  // 0..127
  if (job >= (kV / 16) / 4) return;       // 125 jobs; whole wave exits together
  const int j0 = job * 4;                 // first vocab tile of this wave
  const int mt = blockIdx.y;
  const int col = lane & 15;
  const int hi8 = (lane >> 4) << 3;
  const int m = mt * 16 + col;

  const unsigned short* arow = H2 + (size_t)m * kH;
  v8f acc[4] = {};
  const int nKT = kH >> 5;                // 32
  gemm4(arow,
        Wpk + ((size_t)(j0 + 0) * nKT << 9) + lane * 16,
        Wpk + ((size_t)(j0 + 1) * nKT << 9) + lane * 16,
        Wpk + ((size_t)(j0 + 2) * nKT << 9) + lane * 16,
        Wpk + ((size_t)(j0 + 3) * nKT << 9) + lane * 16,
        nKT, hi8, acc);

#pragma unroll
  for (int u = 0; u < 4; ++u) {
    const int n = (j0 + u) * 16 + col;
    const float bias = bout[n];
#pragma unroll
    for (int r = 0; r < 8; ++r) {
      const int row = mt * 16 + hi8 + r;
      out[((size_t)row * kT + t) * kV + n] = acc[u][r] + bias;
    }
  }
}

// Per-row (batch) max and log-sum-exp for timestep t.
__global__ void rowstats_kernel(const float* __restrict__ out,
                                float* __restrict__ stats, int t) {
  const int m = blockIdx.x;
  const float* row = out + ((size_t)m * kT + t) * kV;
  __shared__ float sm[256];
  float mx = -INFINITY;
  for (int i = threadIdx.x; i < kV; i += 256) mx = fmaxf(mx, row[i]);
  sm[threadIdx.x] = mx;
  __syncthreads();
  for (int s = 128; s > 0; s >>= 1) {
    if (threadIdx.x < s) sm[threadIdx.x] = fmaxf(sm[threadIdx.x], sm[threadIdx.x + s]);
    __syncthreads();
  }
  mx = sm[0];
  __syncthreads();
  float sum = 0.0f;
  for (int i = threadIdx.x; i < kV; i += 256) sum += expf(row[i] - mx);
  sm[threadIdx.x] = sum;
  __syncthreads();
  for (int s = 128; s > 0; s >>= 1) {
    if (threadIdx.x < s) sm[threadIdx.x] += sm[threadIdx.x + s];
    __syncthreads();
  }
  if (threadIdx.x == 0) {
    stats[m * 2 + 0] = mx;
    stats[m * 2 + 1] = logf(sm[0]);
  }
}

// logp = logit - max - log(sum(exp(logit - max)))
__global__ void lsm_norm_kernel(float* __restrict__ out,
                                const float* __restrict__ stats, int t) {
  const int tid = blockIdx.x * blockDim.x + threadIdx.x;
  if (tid >= kB * kV) return;
  const int m = tid / kV;
  const int n = tid - m * kV;
  const size_t off = ((size_t)m * kT + t) * kV + n;
  out[off] -= stats[m * 2] + stats[m * 2 + 1];
}

// ---------------------------------------------------------------------------
// Host launch
// ---------------------------------------------------------------------------
extern "C" void kernel_launch(void* const* d_in, const int* in_sizes, int n_in,
                              void* d_out, int out_size, void* d_ws, size_t ws_size,
                              hipStream_t stream) {
  (void)in_sizes; (void)n_in; (void)out_size; (void)ws_size;

  const int*   gt     = (const int*)d_in[0];
  const float* h0     = (const float*)d_in[1];
  const float* c0     = (const float*)d_in[2];
  const float* emb    = (const float*)d_in[3];
  const float* W_ih1  = (const float*)d_in[4];
  const float* W_hh1  = (const float*)d_in[5];
  const float* b_ih1  = (const float*)d_in[6];
  const float* b_hh1  = (const float*)d_in[7];
  const float* W_ih2  = (const float*)d_in[8];
  const float* W_hh2  = (const float*)d_in[9];
  const float* b_ih2  = (const float*)d_in[10];
  const float* b_hh2  = (const float*)d_in[11];
  const float* W_out  = (const float*)d_in[12];
  const float* b_out  = (const float*)d_in[13];
  float* out = (float*)d_out;

  // Carve workspace (256B-aligned regions)
  unsigned char* wp = (unsigned char*)d_ws;
  auto carve = [&](size_t bytes) -> void* {
    void* p = (void*)wp;
    wp += (bytes + 255) & ~(size_t)255;
    return p;
  };
  unsigned short* emb_bf  = (unsigned short*)carve((size_t)kV * kE * 2);
  unsigned short* wih1_pk = (unsigned short*)carve((size_t)4 * kH * kE * 2);
  unsigned short* whh1_pk = (unsigned short*)carve((size_t)4 * kH * kH * 2);
  unsigned short* wih2_pk = (unsigned short*)carve((size_t)4 * kH * kH * 2);
  unsigned short* whh2_pk = (unsigned short*)carve((size_t)4 * kH * kH * 2);
  unsigned short* wout_pk = (unsigned short*)carve((size_t)kV * kH * 2);
  float* bsum1 = (float*)carve((size_t)4 * kH * 4);
  float* bsum2 = (float*)carve((size_t)4 * kH * 4);
  unsigned short* h1buf = (unsigned short*)carve((size_t)2 * kB * kH * 2); // ping-pong
  unsigned short* h2buf = (unsigned short*)carve((size_t)2 * kB * kH * 2);
  float* c1 = (float*)carve((size_t)kB * kH * 4);
  float* c2 = (float*)carve((size_t)kB * kH * 4);
  float* stats = (float*)carve((size_t)kB * 2 * 4);

  const int BH = kB * kH;

  // --- prep (t-independent, recomputed every call for determinism) ---
  {
    int n = kV * kE;
    cvt_bf16_kernel<<<(n + 255) / 256, 256, 0, stream>>>(emb, emb_bf, n);
  }
  {
    long long n = (long long)4 * kH * kE;
    pack_w_kernel<<<(unsigned)((n + 255) / 256), 256, 0, stream>>>(W_ih1, wih1_pk, 4 * kH, kE);
  }
  {
    long long n = (long long)4 * kH * kH;
    pack_w_kernel<<<(unsigned)((n + 255) / 256), 256, 0, stream>>>(W_hh1, whh1_pk, 4 * kH, kH);
    pack_w_kernel<<<(unsigned)((n + 255) / 256), 256, 0, stream>>>(W_ih2, wih2_pk, 4 * kH, kH);
    pack_w_kernel<<<(unsigned)((n + 255) / 256), 256, 0, stream>>>(W_hh2, whh2_pk, 4 * kH, kH);
  }
  {
    long long n = (long long)kV * kH;
    pack_w_kernel<<<(unsigned)((n + 255) / 256), 256, 0, stream>>>(W_out, wout_pk, kV, kH);
  }
  bsum_kernel<<<(4 * kH + 255) / 256, 256, 0, stream>>>(b_ih1, b_hh1, bsum1, 4 * kH);
  bsum_kernel<<<(4 * kH + 255) / 256, 256, 0, stream>>>(b_ih2, b_hh2, bsum2, 4 * kH);
  init_state_kernel<<<(BH + 255) / 256, 256, 0, stream>>>(h0, c0, h1buf, h2buf, c1, c2, BH);

  // --- sequential timestep loop ---
  for (int t = 0; t < kT; ++t) {
    const int p = t & 1;
    unsigned short* h1_prev = h1buf + (size_t)p * BH;
    unsigned short* h1_new  = h1buf + (size_t)(1 - p) * BH;
    unsigned short* h2_prev = h2buf + (size_t)p * BH;
    unsigned short* h2_new  = h2buf + (size_t)(1 - p) * BH;

    // Layer 1: x_t = emb[gt[:, t]]
    lstm_cell_kernel<true><<<dim3(8, 4), 256, 0, stream>>>(
        emb_bf, gt + t, kE, wih1_pk, h1_prev, kH, whh1_pk, bsum1, c1, h1_new);
    // Layer 2: x = h1_new
    lstm_cell_kernel<false><<<dim3(8, 4), 256, 0, stream>>>(
        h1_new, nullptr, kH, wih2_pk, h2_prev, kH, whh2_pk, bsum2, c2, h2_new);
    // Logits + log-softmax
    logits_kernel<<<dim3(16, 4), 256, 0, stream>>>(h2_new, wout_pk, b_out, out, t);
    rowstats_kernel<<<kB, 256, 0, stream>>>(out, stats, t);
    lsm_norm_kernel<<<(kB * kV + 255) / 256, 256, 0, stream>>>(out, stats, t);
  }
}
